// painn_79869211837099
// MI455X (gfx1250) — matrix operand order repaired and checked by reading, not compiled, not run on hardware
//
#include <hip/hip_runtime.h>
#include <hip/hip_bf16.h>
#include <math.h>

// ---------- types ----------
typedef __attribute__((ext_vector_type(16))) __bf16   v16bf;
typedef __attribute__((ext_vector_type(8)))  float    v8f;
typedef __attribute__((ext_vector_type(4)))  unsigned vu4;

#define NN_NODES 20000
#define NN_EDGES 120000
#define FF       128
#define NG       1000

__device__ __forceinline__ unsigned short bf16rne(float x) {
    unsigned u = __float_as_uint(x);
    u += 0x7FFFu + ((u >> 16) & 1u);
    return (unsigned short)(u >> 16);
}
__device__ __forceinline__ float b2f(unsigned short h) {
    return __uint_as_float(((unsigned)h) << 16);
}
__device__ __forceinline__ float silu_f(float x) {
    // x * sigmoid(x) via v_rcp_f32 (avoids the exact-division sequence)
    return x * __builtin_amdgcn_rcpf(1.f + __expf(-x));
}
__device__ __forceinline__ void atomicAddF(float* p, float v) {
    __hip_atomic_fetch_add(p, v, __ATOMIC_RELAXED, __HIP_MEMORY_SCOPE_AGENT);
}

// ---------- bf16 WMMA GEMM: C[M,N] = A[M,K] @ W[K,N] (+bias)(opt SiLU) ----------
// A: bf16 row-major [M,K]. Wt: bf16 [N,K] (column of W contiguous in K).
// Each wave computes a 32x16 tile (2 WMMAs sharing one B fragment per k-step).
// grid = (M/32, N/128), block = 256 (8 waves -> 8 column tiles).
// K is a compile-time constant so the loop fully unrolls.
template <int K>
__global__ __launch_bounds__(256) void gemm_bf16_wmma(
    const unsigned short* __restrict__ A, const unsigned short* __restrict__ Wt,
    const float* __restrict__ bias, void* __restrict__ C,
    int M, int N, int out_bf16, int fuse_silu)
{
    const int lane  = threadIdx.x & 31;
    const int wave  = threadIdx.x >> 5;
    const int mtile = blockIdx.x << 5;                 // 32 rows per wave
    const int ntile = (blockIdx.y << 7) + (wave << 4);
    if (ntile >= N) return;                            // wave-uniform

    const int rowA = mtile + (lane & 15);
    const int colB = ntile + (lane & 15);
    const int ka   = (lane >> 4) << 3;                 // A half-lane K offset: 0 / 8
    const int kb   = (lane >> 4) << 3;                 // B half-lane K offset: 0 / 8

    const unsigned short* aRow0 = A  + (size_t)rowA * K;          // tile rows 0..15
    const unsigned short* aRow1 = aRow0 + (size_t)16 * K;         // tile rows 16..31
    const unsigned short* bCol  = Wt + (size_t)colB * K;

    v8f acc0 = {0.f,0.f,0.f,0.f,0.f,0.f,0.f,0.f};
    v8f acc1 = {0.f,0.f,0.f,0.f,0.f,0.f,0.f,0.f};

#pragma unroll
    for (int k0 = 0; k0 < K; k0 += 32) {
        // A fragment: lane holds K = ka+{0..7} (v0..3) and ka+16+{0..7} (v4..7)
        union { v16bf v; vu4 q[2]; } a0, a1, b;
        a0.q[0] = *(const vu4*)(aRow0 + k0 + ka);
        a0.q[1] = *(const vu4*)(aRow0 + k0 + ka + 16);
        a1.q[0] = *(const vu4*)(aRow1 + k0 + ka);
        a1.q[1] = *(const vu4*)(aRow1 + k0 + ka + 16);
        // B fragment: lane = column; lane holds K = kb+{0..7} (v0..3) and kb+16+{0..7} (v4..7)
        b.q[0]  = *(const vu4*)(bCol + k0 + kb);
        b.q[1]  = *(const vu4*)(bCol + k0 + kb + 16);

        acc0 = __builtin_amdgcn_wmma_f32_16x16x32_bf16(false, a0.v, false, b.v,
                                                       (short)0, acc0, false, false);
        acc1 = __builtin_amdgcn_wmma_f32_16x16x32_bf16(false, a1.v, false, b.v,
                                                       (short)0, acc1, false, false);
    }

    // epilogue: C layout lanes 0-15 -> M = base+r, lanes 16-31 -> M = base+8+r
    const int   n    = colB;
    const float bval = bias ? bias[n] : 0.f;
    const int   mb0  = mtile + ((lane >> 4) << 3);
#pragma unroll
    for (int r = 0; r < 8; ++r) {
        float x0 = acc0[r] + bval;
        float x1 = acc1[r] + bval;
        if (fuse_silu) { x0 = silu_f(x0); x1 = silu_f(x1); }
        size_t i0 = (size_t)(mb0 + r) * N + n;
        size_t i1 = (size_t)(mb0 + 16 + r) * N + n;
        if (out_bf16) {
            ((unsigned short*)C)[i0] = bf16rne(x0);
            ((unsigned short*)C)[i1] = bf16rne(x1);
        } else {
            ((float*)C)[i0] = x0;
            ((float*)C)[i1] = x1;
        }
    }
}

// ---------- weight pack: Wt[n*Kpad + k] = bf16(W[k*N + n]), zero-pad K ----------
__global__ __launch_bounds__(256) void pack_weight_k(
    const float* __restrict__ W, unsigned short* __restrict__ Wt,
    int K, int N, int Kpad)
{
    int i = blockIdx.x * 256 + threadIdx.x;
    if (i >= N * Kpad) return;
    int n = i / Kpad, k = i - n * Kpad;
    Wt[i] = bf16rne((k < K) ? W[(size_t)k * N + n] : 0.f);
}

// ---------- elementwise helpers ----------
__global__ __launch_bounds__(256) void zero_k(float* __restrict__ p, size_t n) {
    size_t i = (size_t)blockIdx.x * 256 + threadIdx.x;
    if (i < n) p[i] = 0.f;
}
__global__ __launch_bounds__(256) void add_inplace_k(float* __restrict__ d,
                                                     const float* __restrict__ s, size_t n) {
    size_t i = (size_t)blockIdx.x * 256 + threadIdx.x;
    if (i < n) d[i] += s[i];
}
__global__ __launch_bounds__(256) void cvt_bf16_k(const float* __restrict__ src,
                                                  unsigned short* __restrict__ dst, size_t n) {
    size_t i = (size_t)blockIdx.x * 256 + threadIdx.x;
    if (i < n) dst[i] = bf16rne(src[i]);
}

// ---------- s = emb[atomic_numbers] ----------
__global__ __launch_bounds__(256) void embed_k(const int* __restrict__ an,
                                               const float* __restrict__ emb,
                                               float* __restrict__ s) {
    int i = blockIdx.x * 256 + threadIdx.x;
    if (i >= NN_NODES * FF) return;
    int n = i >> 7, f = i & 127;
    s[i] = emb[an[n] * FF + f];
}

// ---------- edge geometry: r, padded rbf (20 -> 32, bf16), global sum(r^2) ----------
__global__ __launch_bounds__(256) void edge_geom_k(
    const float* __restrict__ pos, const int* __restrict__ ii, const int* __restrict__ jj,
    float* __restrict__ r, unsigned short* __restrict__ rbfp,
    float* __restrict__ gnorm, int E)
{
    int e = blockIdx.x * 256 + threadIdx.x;
    float ss = 0.f;
    if (e < E) {
        int i = ii[e], j = jj[e];
        float rx = pos[3 * i]     - pos[3 * j];
        float ry = pos[3 * i + 1] - pos[3 * j + 1];
        float rz = pos[3 * i + 2] - pos[3 * j + 2];
        r[3 * (size_t)e] = rx; r[3 * (size_t)e + 1] = ry; r[3 * (size_t)e + 2] = rz;
        float d2 = rx * rx + ry * ry + rz * rz;
        ss = d2;
        float rn = sqrtf(d2);
        const float c1 = 3.14159265358979323846f * 0.5f;   // pi / R_CUT, R_CUT = 2
#pragma unroll
        for (int k = 0; k < 32; ++k) {
            float val = 0.f;
            if (k < 20) {
                float t = sinf((float)(k + 1) * c1 * rn) / rn;
                val = (t <= 2.0f) ? 0.5f * (cosf(c1 * t) + 1.f) : 0.f;
            }
            rbfp[(size_t)e * 32 + k] = bf16rne(val);
        }
    }
    __shared__ float red[256];
    red[threadIdx.x] = ss;
    __syncthreads();
    for (int st = 128; st > 0; st >>= 1) {
        if (threadIdx.x < st) red[threadIdx.x] += red[threadIdx.x + st];
        __syncthreads();
    }
    if (threadIdx.x == 0) atomicAddF(&gnorm[0], red[0]);
}

__global__ void gnorm_fin_k(float* gnorm) {
    gnorm[1] = 1.f / sqrtf(gnorm[0]);   // reference: r_dir = r / ||r||_Frobenius (global!)
}

// ---------- edge message: split = w_edge * phi[idx_i]; scatter-add (bf16 inputs) ----------
__global__ __launch_bounds__(256) void edge_message_k(
    const int* __restrict__ ii, const unsigned short* __restrict__ wE,
    const unsigned short* __restrict__ phi,
    const float* __restrict__ v, const float* __restrict__ r, const float* __restrict__ gnorm,
    float* __restrict__ accS, float* __restrict__ acc3, int E)
{
    int idx = blockIdx.x * 256 + threadIdx.x;
    if (idx >= E * FF) return;
    int e = idx >> 7, f = idx & 127;
    int i = ii[e];
    size_t we = (size_t)e * 384, pe = (size_t)i * 384;
    float m0 = b2f(wE[we + f])       * b2f(phi[pe + f]);
    float m1 = b2f(wE[we + 128 + f]) * b2f(phi[pe + 128 + f]);
    float m2 = b2f(wE[we + 256 + f]) * b2f(phi[pe + 256 + f]);
    atomicAddF(&accS[(size_t)i * FF + f], m1);
    float inv = gnorm[1];
#pragma unroll
    for (int x = 0; x < 3; ++x) {
        float ve = m0 * v[((size_t)i * 3 + x) * FF + f] + m2 * (r[3 * (size_t)e + x] * inv);
        atomicAddF(&acc3[((size_t)i * 3 + x) * FF + f], ve);
    }
}

// ---------- s_stack = [ ||vV||_x , s ]  (N x 256, bf16 out) ----------
__global__ __launch_bounds__(256) void make_sstack_k(
    const float* __restrict__ vV, const float* __restrict__ s,
    unsigned short* __restrict__ sst)
{
    int i = blockIdx.x * 256 + threadIdx.x;
    if (i >= NN_NODES * FF) return;
    int n = i >> 7, f = i & 127;
    float a0 = vV[((size_t)n * 3 + 0) * FF + f];
    float a1 = vV[((size_t)n * 3 + 1) * FF + f];
    float a2 = vV[((size_t)n * 3 + 2) * FF + f];
    sst[(size_t)n * 256 + f]       = bf16rne(sqrtf(a0 * a0 + a1 * a1 + a2 * a2));
    sst[(size_t)n * 256 + 128 + f] = bf16rne(s[(size_t)n * FF + f]);
}

// ---------- update combine: v += u*a0 ; s += sum_x(u*vV)*a1 + a2 ----------
__global__ __launch_bounds__(256) void update_combine_k(
    const float* __restrict__ a, const float* __restrict__ u, const float* __restrict__ vV,
    float* __restrict__ v, float* __restrict__ s)
{
    int i = blockIdx.x * 256 + threadIdx.x;
    if (i >= NN_NODES * FF) return;
    int n = i >> 7, f = i & 127;
    float a0 = a[(size_t)n * 384 + f];
    float a1 = a[(size_t)n * 384 + 128 + f];
    float a2 = a[(size_t)n * 384 + 256 + f];
    float dot = 0.f;
#pragma unroll
    for (int x = 0; x < 3; ++x) {
        size_t ix = ((size_t)n * 3 + x) * FF + f;
        float uu = u[ix];
        dot += uu * vV[ix];
        v[ix] += uu * a0;
    }
    s[(size_t)n * FF + f] += dot * a1 + a2;
}

// ---------- final: per-node feature sum -> segment_sum over graphs ----------
__global__ __launch_bounds__(128) void final_reduce_k(
    const float* __restrict__ of, const int* __restrict__ gidx, float* __restrict__ out)
{
    int n = blockIdx.x;
    __shared__ float red[128];
    red[threadIdx.x] = of[(size_t)n * FF + threadIdx.x];
    __syncthreads();
    for (int st = 64; st > 0; st >>= 1) {
        if (threadIdx.x < st) red[threadIdx.x] += red[threadIdx.x + st];
        __syncthreads();
    }
    if (threadIdx.x == 0) atomicAddF(&out[gidx[n]], red[0]);
}

// ---------- host-side GEMM launcher ----------
template <int K>
static void launch_gemm(const unsigned short* A, const unsigned short* Wt, const float* bias,
                        void* C, int M, int N, int out_bf16, int silu, hipStream_t stream)
{
    dim3 g(M / 32, N / 128);
    gemm_bf16_wmma<K><<<g, 256, 0, stream>>>(A, Wt, bias, C, M, N, out_bf16, silu);
}

// =====================================================================
extern "C" void kernel_launch(void* const* d_in, const int* in_sizes, int n_in,
                              void* d_out, int out_size, void* d_ws, size_t ws_size,
                              hipStream_t stream)
{
    (void)in_sizes; (void)n_in; (void)out_size; (void)ws_size;
    const int NN = NN_NODES, NE = NN_EDGES;

    const int*   an    = (const int*)  d_in[0];
    const float* pos   = (const float*)d_in[1];
    const int*   gidx  = (const int*)  d_in[2];
    const int*   idx_i = (const int*)  d_in[3];
    const int*   idx_j = (const int*)  d_in[4];
    const float* emb   = (const float*)d_in[5];
    const float* W_phi1 = (const float*)d_in[6];  const float* b_phi1 = (const float*)d_in[7];
    const float* W_phi2 = (const float*)d_in[8];  const float* b_phi2 = (const float*)d_in[9];
    const float* W_w    = (const float*)d_in[10]; const float* b_w    = (const float*)d_in[11];
    const float* W_a1   = (const float*)d_in[12]; const float* b_a1   = (const float*)d_in[13];
    const float* W_a2   = (const float*)d_in[14]; const float* b_a2   = (const float*)d_in[15];
    const float* W_V    = (const float*)d_in[16];
    const float* W_U    = (const float*)d_in[17];
    const float* W_o1   = (const float*)d_in[18]; const float* b_o1   = (const float*)d_in[19];
    const float* W_o2   = (const float*)d_in[20]; const float* b_o2   = (const float*)d_in[21];

    // ---- workspace layout ----
    char* base = (char*)d_ws;
    size_t off = 0;
    auto allocb = [&](size_t nbytes) -> void* {
        void* p = (void*)(base + off);
        off += ((nbytes + 255) & ~(size_t)255);
        return p;
    };
    // f32 state + accumulators
    float* s     = (float*)allocb((size_t)NN * FF * 4);
    float* v     = (float*)allocb((size_t)NN * 3 * FF * 4);
    float* rbuf  = (float*)allocb((size_t)NE * 3 * 4);
    float* acc3  = (float*)allocb((size_t)NN * 3 * FF * 4);   // dv (msg) / vV (upd)
    float* accS  = (float*)allocb((size_t)NN * FF * 4);       // ds (msg)
    float* ubuf  = (float*)allocb((size_t)NN * 3 * FF * 4);   // u (upd)
    float* aF    = (float*)allocb((size_t)NN * 384 * 4);      // a (upd)
    float* ofbuf = (float*)allocb((size_t)NN * FF * 4);       // head features
    float* gnorm = (float*)allocb(256);                       // [0]=sum r^2, [1]=1/sqrt
    // bf16 activations (GEMM inputs/outputs)
    unsigned short* s_bf    = (unsigned short*)allocb((size_t)NN * FF * 2);
    unsigned short* v_bf    = (unsigned short*)allocb((size_t)NN * 3 * FF * 2);
    unsigned short* vv_bf   = (unsigned short*)allocb((size_t)NN * 3 * FF * 2);
    unsigned short* rbfp_bf = (unsigned short*)allocb((size_t)NE * 32 * 2);
    unsigned short* wE_bf   = (unsigned short*)allocb((size_t)NE * 384 * 2);
    unsigned short* phi_bf  = (unsigned short*)allocb((size_t)NN * 384 * 2);
    unsigned short* tmpH_bf = (unsigned short*)allocb((size_t)NN * 256 * 2); // h / sstack
    unsigned short* hbuf_bf = (unsigned short*)allocb((size_t)NN * FF * 2);  // ha / head h
    // bf16 packed weights [N,K]
    unsigned short* Wphi1t = (unsigned short*)allocb((size_t)128 * 128 * 2);
    unsigned short* Wphi2t = (unsigned short*)allocb((size_t)384 * 128 * 2);
    unsigned short* Wwt    = (unsigned short*)allocb((size_t)384 * 32 * 2);
    unsigned short* Wa1t   = (unsigned short*)allocb((size_t)128 * 256 * 2);
    unsigned short* Wa2t   = (unsigned short*)allocb((size_t)384 * 128 * 2);
    unsigned short* WVt    = (unsigned short*)allocb((size_t)128 * 128 * 2);
    unsigned short* WUt    = (unsigned short*)allocb((size_t)128 * 128 * 2);
    unsigned short* Wo1t   = (unsigned short*)allocb((size_t)128 * 128 * 2);
    unsigned short* Wo2t   = (unsigned short*)allocb((size_t)128 * 128 * 2);

    auto zero = [&](float* p, size_t n) {
        zero_k<<<(unsigned)((n + 255) / 256), 256, 0, stream>>>(p, n);
    };
    auto cvt = [&](const float* src, unsigned short* dst, size_t n) {
        cvt_bf16_k<<<(unsigned)((n + 255) / 256), 256, 0, stream>>>(src, dst, n);
    };
    auto pack = [&](const float* W, unsigned short* Wt, int K, int N, int Kpad) {
        int tot = N * Kpad;
        pack_weight_k<<<(tot + 255) / 256, 256, 0, stream>>>(W, Wt, K, N, Kpad);
    };

    // ---- one-time setup ----
    pack(W_phi1, Wphi1t, 128, 128, 128);
    pack(W_phi2, Wphi2t, 128, 384, 128);
    pack(W_w,    Wwt,     20, 384,  32);
    pack(W_a1,   Wa1t,   256, 128, 256);
    pack(W_a2,   Wa2t,   128, 384, 128);
    pack(W_V,    WVt,    128, 128, 128);
    pack(W_U,    WUt,    128, 128, 128);
    pack(W_o1,   Wo1t,   128, 128, 128);
    pack(W_o2,   Wo2t,   128, 128, 128);

    zero(gnorm, 8);
    embed_k<<<(NN * FF + 255) / 256, 256, 0, stream>>>(an, emb, s);
    zero(v, (size_t)NN * 3 * FF);
    edge_geom_k<<<(NE + 255) / 256, 256, 0, stream>>>(pos, idx_i, idx_j,
                                                      rbuf, rbfp_bf, gnorm, NE);
    gnorm_fin_k<<<1, 1, 0, stream>>>(gnorm);
    // w_edge = rbf @ W_w + b_w (loop-invariant; stored bf16 -> halves stream traffic)
    launch_gemm<32>(rbfp_bf, Wwt, b_w, wE_bf, NE, 384, /*bf16*/1, 0, stream);

    // ---- 3 x (message, update) ----
    for (int it = 0; it < 3; ++it) {
        // phi computed PER NODE (6x less GEMM work than the reference's per-edge form)
        cvt(s, s_bf, (size_t)NN * FF);
        launch_gemm<128>(s_bf, Wphi1t, b_phi1, tmpH_bf, NN, 128, 1, /*silu*/1, stream);
        launch_gemm<128>(tmpH_bf, Wphi2t, b_phi2, phi_bf, NN, 384, 1, 0, stream);

        zero(acc3, (size_t)NN * 3 * FF);
        zero(accS, (size_t)NN * FF);
        edge_message_k<<<(NE * FF) / 256, 256, 0, stream>>>(idx_i, wE_bf, phi_bf, v, rbuf,
                                                            gnorm, accS, acc3, NE);
        add_inplace_k<<<((size_t)NN * 3 * FF + 255) / 256, 256, 0, stream>>>(v, acc3, (size_t)NN * 3 * FF);
        add_inplace_k<<<((size_t)NN * FF + 255) / 256, 256, 0, stream>>>(s, accS, (size_t)NN * FF);

        // update
        cvt(v, v_bf, (size_t)NN * 3 * FF);
        launch_gemm<128>(v_bf, WVt, nullptr, acc3, NN * 3, 128, 0, 0, stream);  // vV (f32)
        cvt(acc3, vv_bf, (size_t)NN * 3 * FF);
        launch_gemm<128>(vv_bf, WUt, nullptr, ubuf, NN * 3, 128, 0, 0, stream); // u  (f32)
        make_sstack_k<<<(NN * FF + 255) / 256, 256, 0, stream>>>(acc3, s, tmpH_bf);
        launch_gemm<256>(tmpH_bf, Wa1t, b_a1, hbuf_bf, NN, 128, 1, 1, stream);  // ha (bf16)
        launch_gemm<128>(hbuf_bf, Wa2t, b_a2, aF, NN, 384, 0, 0, stream);       // a  (f32)
        update_combine_k<<<(NN * FF + 255) / 256, 256, 0, stream>>>(aF, ubuf, acc3, v, s);
    }

    // ---- output head + graph segment sum ----
    cvt(s, s_bf, (size_t)NN * FF);
    launch_gemm<128>(s_bf, Wo1t, b_o1, hbuf_bf, NN, 128, 1, 1, stream);
    launch_gemm<128>(hbuf_bf, Wo2t, b_o2, ofbuf, NN, 128, 0, 0, stream);
    zero((float*)d_out, NG);
    final_reduce_k<<<NN, 128, 0, stream>>>(ofbuf, gidx, (float*)d_out);
}
